// MultiHeadAttention_34265249087840
// MI455X (gfx1250) — matrix operand order repaired and checked by reading, not compiled, not run on hardware
//
#include <hip/hip_runtime.h>
#include <hip/hip_bf16.h>

// B=2, T=2048, D=512, H=8, HD=64
#define B_SZ   2
#define T_SEQ  2048
#define D_MODEL 512
#define N_HEADS 8
#define HDIM   64

typedef __bf16 bf16_t;
typedef __attribute__((ext_vector_type(16))) __bf16 v16bf;
typedef __attribute__((ext_vector_type(8)))  __bf16 v8bf;
typedef __attribute__((ext_vector_type(8)))  float  v8f;

static __device__ __forceinline__ v8f zero8() { v8f z = {}; return z; }

static __device__ __forceinline__ bf16_t f2bf(float f) {
  unsigned u = __float_as_uint(f);
  unsigned r = (u + 0x7FFFu + ((u >> 16) & 1u)) >> 16;  // RNE
  unsigned short s = (unsigned short)r;
  return __builtin_bit_cast(bf16_t, s);
}
static __device__ __forceinline__ float bf2f(bf16_t b) {
  unsigned short s = __builtin_bit_cast(unsigned short, b);
  return __uint_as_float(((unsigned)s) << 16);
}

static __device__ __forceinline__ v8f wmma_bf(v16bf a, v16bf b, v8f c) {
  // D(f32 16x16) = A(bf16 16x32) * B(bf16 32x16) + C
  return __builtin_amdgcn_wmma_f32_16x16x32_bf16(false, a, false, b, (short)0, c,
                                                 false, false);
}

// A-fragment (16x32 bf16). Per lane: M = lane%16 (rowptr points at that row),
// K = {kbase + hf*8 .. +7} U {kbase+16 + hf*8 .. +7}, hf = lane/16.
static __device__ __forceinline__ v16bf load_a_frag(const bf16_t* rowptr, int kbase, int hf) {
  union { v16bf v; v8bf h[2]; } u;
  const bf16_t* p = rowptr + kbase + hf * 8;
  u.h[0] = *(const v8bf*)(p);
  u.h[1] = *(const v8bf*)(p + 16);
  return u.v;
}
// B-fragment (32x16 bf16). Per lane: N fixed (colptr = base + N*stride),
// K = {kbase + hf*8 .. +7} U {kbase+16 + hf*8 .. +7}, hf = lane/16.
static __device__ __forceinline__ v16bf load_b_frag(const bf16_t* colptr, int kbase, int hf) {
  union { v16bf v; v8bf h[2]; } u;
  const bf16_t* p = colptr + kbase + hf * 8;
  u.h[0] = *(const v8bf*)(p);
  u.h[1] = *(const v8bf*)(p + 16);
  return u.v;
}

// ---------------------------------------------------------------------------
// Kernel 1: rel_emb fp32 -> bf16
// ---------------------------------------------------------------------------
__global__ void rel_cvt_kernel(const float* __restrict__ rel, bf16_t* __restrict__ relbf, int n) {
  int i = blockIdx.x * blockDim.x + threadIdx.x;
  if (i < n) relbf[i] = f2bf(rel[i]);
}

// ---------------------------------------------------------------------------
// Kernel 2: qkv = x @ Wqkv, scattered to q[bh][t][d], k[bh][t][d] (*0.125),
//           v[bh][d][t] (transposed), all bf16.
// ---------------------------------------------------------------------------
__global__ __launch_bounds__(256) void qkv_gemm_kernel(
    const float* __restrict__ x,      // [4096][512]
    const float* __restrict__ w,      // [512][1536]
    bf16_t* __restrict__ qbf, bf16_t* __restrict__ kbf, bf16_t* __restrict__ vbf) {
  __shared__ bf16_t Xlds[128 * 40];   // [row][k], stride 40 (pad)
  __shared__ bf16_t Wlds[128 * 40];   // [n][k]  (W transposed), stride 40

  const int tid = threadIdx.x;
  const int lane = tid & 31, wave = tid >> 5;
  const int ln = lane & 15, hf = lane >> 4;
  const int wm = wave & 3, wn = wave >> 2;
  const int R0 = blockIdx.x * 128;
  const int C0 = blockIdx.y * 128;

  v8f acc[2][4];
#pragma unroll
  for (int i = 0; i < 2; i++)
#pragma unroll
    for (int j = 0; j < 4; j++) acc[i][j] = zero8();

  for (int kb = 0; kb < 512; kb += 32) {
    __syncthreads();
    for (int i = tid; i < 1024; i += 256) {           // X tile 128x32 f32->bf16
      int r = i >> 3, kq = (i & 7) << 2;
      float4 f = *(const float4*)(x + (size_t)(R0 + r) * 512 + kb + kq);
      bf16_t* dst = Xlds + r * 40 + kq;
      dst[0] = f2bf(f.x); dst[1] = f2bf(f.y); dst[2] = f2bf(f.z); dst[3] = f2bf(f.w);
    }
    {                                                 // W tile transposed
      int kk = tid >> 3, ng = tid & 7;
      const float* src = w + (size_t)(kb + kk) * 1536 + C0 + ng * 16;
#pragma unroll
      for (int q = 0; q < 4; q++) {
        float4 f = *(const float4*)(src + q * 4);
        int n = ng * 16 + q * 4;
        Wlds[(n + 0) * 40 + kk] = f2bf(f.x);
        Wlds[(n + 1) * 40 + kk] = f2bf(f.y);
        Wlds[(n + 2) * 40 + kk] = f2bf(f.z);
        Wlds[(n + 3) * 40 + kk] = f2bf(f.w);
      }
    }
    __syncthreads();
    v16bf a0 = load_a_frag(Xlds + (size_t)(wm * 32 + ln) * 40, 0, hf);
    v16bf a1 = load_a_frag(Xlds + (size_t)(wm * 32 + 16 + ln) * 40, 0, hf);
#pragma unroll
    for (int nt = 0; nt < 4; nt++) {
      v16bf b = load_b_frag(Wlds + (size_t)(wn * 64 + nt * 16 + ln) * 40, 0, hf);
      acc[0][nt] = wmma_bf(a0, b, acc[0][nt]);
      acc[1][nt] = wmma_bf(a1, b, acc[1][nt]);
    }
  }
  // Epilogue: C layout -> q/k/v scatter.
#pragma unroll
  for (int mt = 0; mt < 2; mt++)
#pragma unroll
    for (int nt = 0; nt < 4; nt++)
#pragma unroll
      for (int v = 0; v < 8; v++) {
        int row = R0 + wm * 32 + mt * 16 + v + hf * 8;
        int c = C0 + wn * 64 + nt * 16 + ln;
        int qk = c >> 9, h = (c >> 6) & 7, d = c & 63;
        int bb = row >> 11, t = row & 2047;
        size_t bhi = (size_t)(bb * N_HEADS + h);
        float val = acc[mt][nt][v];
        if (qk == 0)      qbf[(bhi * T_SEQ + t) * HDIM + d] = f2bf(val);
        else if (qk == 1) kbf[(bhi * T_SEQ + t) * HDIM + d] = f2bf(val * 0.125f);
        else              vbf[(bhi * HDIM + d) * T_SEQ + t] = f2bf(val);
      }
}

// ---------------------------------------------------------------------------
// Kernel 3: flash attention with relative-position skew.
// Grid: (B*H, T/64). Block: 128 threads = 4 waves; wave owns 16 query rows.
// logits[i,j] = q_i.k'_j + q_i.rel[(T-1)+j-i]  (k pre-scaled by 1/8)
// K/V/rel tiles are staged into LDS once per column tile (shared by the 4
// waves); P and G scratch are wave-private. Dynamic LDS = 106,496 B.
// ---------------------------------------------------------------------------
#define KSTR 72    // Klds/Elds row stride (36 dwords, gcd(36,64)=4 -> no conflicts)
#define VSTR 136   // Vlds row stride (68 dwords)
#define PSTR 136   // P row stride
#define GSTR 200   // G row stride

__global__ __launch_bounds__(128) void attn_rel_kernel(
    const bf16_t* __restrict__ qbf,   // [BH][T][64]
    const bf16_t* __restrict__ kbf,   // [BH][T][64]
    const bf16_t* __restrict__ vbf,   // [BH][64][T]
    const bf16_t* __restrict__ relbf, // [4095][64]
    bf16_t* __restrict__ obf)         // [B*T][512]
{
  extern __shared__ bf16_t smem[];
  bf16_t* Klds = smem;                       // [128][KSTR]
  bf16_t* Vlds = Klds + 128 * KSTR;          // [64][VSTR]   (V^T: [d][j])
  bf16_t* Elds = Vlds + 64 * VSTR;           // [192][KSTR]
  bf16_t* Plds = Elds + 192 * KSTR;          // [4][16][PSTR] per-wave
  bf16_t* Glds = Plds + 4 * 16 * PSTR;       // [4][16][GSTR] per-wave

  const int tid = threadIdx.x;
  const int lane = tid & 31;
  const int wave = tid >> 5;
  const int ln = lane & 15;
  const int hf = lane >> 4;

  const int bh = blockIdx.x;
  const int I0 = blockIdx.y * 64;

  const bf16_t* qg = qbf + (size_t)bh * T_SEQ * HDIM;
  const bf16_t* kg = kbf + (size_t)bh * T_SEQ * HDIM;
  const bf16_t* vg = vbf + (size_t)bh * HDIM * T_SEQ;

  // Q fragments (A layout), loaded once, reused for all column tiles.
  const bf16_t* qrow = qg + (size_t)(I0 + wave * 16 + ln) * HDIM;
  v16bf qfrag0 = load_a_frag(qrow, 0, hf);
  v16bf qfrag1 = load_a_frag(qrow, 32, hf);

  v8f Oacc[4];
  float rowmax[8], rowsum[8];
#pragma unroll
  for (int i = 0; i < 4; i++) Oacc[i] = zero8();
#pragma unroll
  for (int i = 0; i < 8; i++) { rowmax[i] = -3.0e38f; rowsum[i] = 0.0f; }

  bf16_t* Pw = Plds + wave * 16 * PSTR;
  bf16_t* Gw = Glds + wave * 16 * GSTR;
  const int Cbase = T_SEQ - 1 - I0 - 63;  // rel row at u=0 for J0=0

  for (int J0 = 0; J0 < T_SEQ; J0 += 128) {
    if (J0 + 128 < T_SEQ) {
      __builtin_prefetch(kg + (size_t)(J0 + 128 + tid) * HDIM, 0, 1);
      __builtin_prefetch(vg + (size_t)(tid >> 1) * T_SEQ + J0 + 128 + (tid & 1) * 64, 0, 1);
    }

    __syncthreads();  // previous iteration's LDS reads complete
    // ---- stage K tile [j][d] ----
    for (int i = tid; i < 1024; i += 128) {
      int r = i >> 3, ch = i & 7;
      *(uint4*)(Klds + r * KSTR + ch * 8) =
          *(const uint4*)(kg + (size_t)(J0 + r) * HDIM + ch * 8);
    }
    // ---- stage V^T tile [d][j] ----
    for (int i = tid; i < 1024; i += 128) {
      int d = i >> 4, ch = i & 15;
      *(uint4*)(Vlds + d * VSTR + ch * 8) =
          *(const uint4*)(vg + (size_t)d * T_SEQ + J0 + ch * 8);
    }
    // ---- stage rel band rows u in [0,192): r = Cbase + J0 + u (clamped) ----
    for (int i = tid; i < 1536; i += 128) {
      int u = i >> 3, ch = i & 7;
      int r = Cbase + J0 + u;
      r = r < 0 ? 0 : (r > 4094 ? 4094 : r);
      *(uint4*)(Elds + u * KSTR + ch * 8) =
          *(const uint4*)(relbf + (size_t)r * HDIM + ch * 8);
    }
    __syncthreads();

    // ---- G[m][u] = q_(wave*16+m) . rel[Cbase+J0+u], u in [0,192) ----
#pragma unroll
    for (int ut = 0; ut < 12; ut++) {
      const bf16_t* ecol = Elds + (size_t)(ut * 16 + ln) * KSTR;
      v8f g = zero8();
      g = wmma_bf(qfrag0, load_b_frag(ecol, 0, hf), g);
      g = wmma_bf(qfrag1, load_b_frag(ecol, 32, hf), g);
#pragma unroll
      for (int v = 0; v < 8; v++)
        Gw[(v + hf * 8) * GSTR + ut * 16 + ln] = f2bf(g[v]);
    }

    // ---- S = Q K'^T ----
    v8f S[8];
#pragma unroll
    for (int nt = 0; nt < 8; nt++) {
      const bf16_t* kcol = Klds + (size_t)(nt * 16 + ln) * KSTR;
      v8f s = zero8();
      s = wmma_bf(qfrag0, load_b_frag(kcol, 0, hf), s);
      s = wmma_bf(qfrag1, load_b_frag(kcol, 32, hf), s);
      S[nt] = s;
    }

    // ---- skew gather: S[i,j] += G[m][ j - i + 63 ] ----
#pragma unroll
    for (int nt = 0; nt < 8; nt++)
#pragma unroll
      for (int v = 0; v < 8; v++) {
        int m = v + hf * 8;
        int u = nt * 16 + ln - (wave * 16 + m) + 63;
        S[nt][v] += bf2f(Gw[m * GSTR + u]);
      }

    // ---- online softmax (rows live in 16-lane groups) ----
    float tmax[8], corr[8];
#pragma unroll
    for (int v = 0; v < 8; v++) {
      float t = S[0][v];
#pragma unroll
      for (int nt = 1; nt < 8; nt++) t = fmaxf(t, S[nt][v]);
#pragma unroll
      for (int off = 1; off < 16; off <<= 1)
        t = fmaxf(t, __shfl_xor(t, off, 32));
      tmax[v] = fmaxf(t, rowmax[v]);
      corr[v] = __expf(rowmax[v] - tmax[v]);
      rowmax[v] = tmax[v];
    }
#pragma unroll
    for (int v = 0; v < 8; v++) {
      float ssum = 0.0f;
#pragma unroll
      for (int nt = 0; nt < 8; nt++) {
        float p = __expf(S[nt][v] - tmax[v]);
        S[nt][v] = p;
        ssum += p;
      }
#pragma unroll
      for (int off = 1; off < 16; off <<= 1)
        ssum += __shfl_xor(ssum, off, 32);
      rowsum[v] = rowsum[v] * corr[v] + ssum;
    }
#pragma unroll
    for (int dt = 0; dt < 4; dt++)
#pragma unroll
      for (int v = 0; v < 8; v++) Oacc[dt][v] *= corr[v];

    // ---- P -> LDS (re-layout C-format -> A-format source) ----
#pragma unroll
    for (int nt = 0; nt < 8; nt++)
#pragma unroll
      for (int v = 0; v < 8; v++)
        Pw[(v + hf * 8) * PSTR + nt * 16 + ln] = f2bf(S[nt][v]);

    // ---- O += P @ V ----
#pragma unroll
    for (int ks = 0; ks < 4; ks++) {
      v16bf pa = load_a_frag(Pw + (size_t)ln * PSTR, ks * 32, hf);
#pragma unroll
      for (int dt = 0; dt < 4; dt++) {
        const bf16_t* vcol = Vlds + (size_t)(dt * 16 + ln) * VSTR;
        Oacc[dt] = wmma_bf(pa, load_b_frag(vcol, ks * 32, hf), Oacc[dt]);
      }
    }
  }

  // ---- epilogue: normalize, write (b,t,h*64+d) ----
  const int bb = bh >> 3, hh = bh & 7;
#pragma unroll
  for (int dt = 0; dt < 4; dt++)
#pragma unroll
    for (int v = 0; v < 8; v++) {
      int m = v + hf * 8;
      int rowg = bb * T_SEQ + I0 + wave * 16 + m;
      int col = hh * 64 + dt * 16 + ln;
      obf[(size_t)rowg * D_MODEL + col] = f2bf(Oacc[dt][v] / rowsum[v]);
    }
}

// ---------------------------------------------------------------------------
// Kernel 4: out = attn_out @ Wproj + bproj (fp32 output)
// ---------------------------------------------------------------------------
__global__ __launch_bounds__(256) void proj_gemm_kernel(
    const bf16_t* __restrict__ obf,   // [4096][512] bf16
    const float* __restrict__ w,      // [512][512]
    const float* __restrict__ bias,   // [512]
    float* __restrict__ out) {        // [4096][512]
  __shared__ bf16_t Xlds[128 * 40];
  __shared__ bf16_t Wlds[128 * 40];

  const int tid = threadIdx.x;
  const int lane = tid & 31, wave = tid >> 5;
  const int ln = lane & 15, hf = lane >> 4;
  const int wm = wave & 3, wn = wave >> 2;
  const int R0 = blockIdx.x * 128;
  const int C0 = blockIdx.y * 128;

  v8f acc[2][4];
#pragma unroll
  for (int i = 0; i < 2; i++)
#pragma unroll
    for (int j = 0; j < 4; j++) acc[i][j] = zero8();

  for (int kb = 0; kb < 512; kb += 32) {
    __syncthreads();
    for (int i = tid; i < 512; i += 256) {            // X tile (already bf16)
      int r = i >> 2, ch = i & 3;
      *(uint4*)(Xlds + r * 40 + ch * 8) =
          *(const uint4*)(obf + (size_t)(R0 + r) * 512 + kb + ch * 8);
    }
    {                                                 // W tile transposed
      int kk = tid >> 3, ng = tid & 7;
      const float* src = w + (size_t)(kb + kk) * 512 + C0 + ng * 16;
#pragma unroll
      for (int q = 0; q < 4; q++) {
        float4 f = *(const float4*)(src + q * 4);
        int n = ng * 16 + q * 4;
        Wlds[(n + 0) * 40 + kk] = f2bf(f.x);
        Wlds[(n + 1) * 40 + kk] = f2bf(f.y);
        Wlds[(n + 2) * 40 + kk] = f2bf(f.z);
        Wlds[(n + 3) * 40 + kk] = f2bf(f.w);
      }
    }
    __syncthreads();
    v16bf a0 = load_a_frag(Xlds + (size_t)(wm * 32 + ln) * 40, 0, hf);
    v16bf a1 = load_a_frag(Xlds + (size_t)(wm * 32 + 16 + ln) * 40, 0, hf);
#pragma unroll
    for (int nt = 0; nt < 4; nt++) {
      v16bf b = load_b_frag(Wlds + (size_t)(wn * 64 + nt * 16 + ln) * 40, 0, hf);
      acc[0][nt] = wmma_bf(a0, b, acc[0][nt]);
      acc[1][nt] = wmma_bf(a1, b, acc[1][nt]);
    }
  }
#pragma unroll
  for (int mt = 0; mt < 2; mt++)
#pragma unroll
    for (int nt = 0; nt < 4; nt++)
#pragma unroll
      for (int v = 0; v < 8; v++) {
        int row = R0 + wm * 32 + mt * 16 + v + hf * 8;
        int c = C0 + wn * 64 + nt * 16 + ln;
        out[(size_t)row * 512 + c] = acc[mt][nt][v] + bias[c];
      }
}

// ---------------------------------------------------------------------------
extern "C" void kernel_launch(void* const* d_in, const int* in_sizes, int n_in,
                              void* d_out, int out_size, void* d_ws, size_t ws_size,
                              hipStream_t stream) {
  const float* x     = (const float*)d_in[0];
  const float* Wqkv  = (const float*)d_in[1];
  const float* Wproj = (const float*)d_in[2];
  const float* bproj = (const float*)d_in[3];
  const float* rel   = (const float*)d_in[4];
  float* out = (float*)d_out;

  const size_t nper = (size_t)B_SZ * N_HEADS * T_SEQ * HDIM;  // 2,097,152
  bf16_t* qbf   = (bf16_t*)d_ws;
  bf16_t* kbf   = qbf + nper;
  bf16_t* vbf   = kbf + nper;
  bf16_t* obf   = vbf + nper;
  bf16_t* relbf = obf + nper;

  const int attn_lds =
      (128 * KSTR + 64 * VSTR + 192 * KSTR + 4 * 16 * PSTR + 4 * 16 * GSTR) *
      (int)sizeof(bf16_t);  // 106,496 bytes
  (void)hipFuncSetAttribute((const void*)attn_rel_kernel,
                            hipFuncAttributeMaxDynamicSharedMemorySize, attn_lds);

  rel_cvt_kernel<<<dim3((4095 * HDIM + 255) / 256), dim3(256), 0, stream>>>(
      rel, relbf, 4095 * HDIM);
  qkv_gemm_kernel<<<dim3(32, 12), dim3(256), 0, stream>>>(x, Wqkv, qbf, kbf, vbf);
  attn_rel_kernel<<<dim3(B_SZ * N_HEADS, T_SEQ / 64), dim3(128), attn_lds, stream>>>(
      qbf, kbf, vbf, relbf, obf);
  proj_gemm_kernel<<<dim3(32, 4), dim3(256), 0, stream>>>(obf, Wproj, bproj, out);
}
